// Model_1769526526664
// MI455X (gfx1250) — matrix operand order repaired and checked by reading, not compiled, not run on hardware
//
#include <hip/hip_runtime.h>
#include <hip/hip_bf16.h>

// ---------------------------------------------------------------------------
// PointNet++ classification forward for MI455X (gfx1250, wave32, WMMA).
// GEMMs: v_wmma_f32_16x16x32_f16, weights staged block-wide into LDS via
// global_load_async_to_lds_b128 (ASYNCcnt), A streamed with prefetch and
// reused across NT=4 column tiles per wave.
// ---------------------------------------------------------------------------

typedef _Float16 half_t;
typedef __attribute__((ext_vector_type(16))) _Float16 v16h;
typedef __attribute__((ext_vector_type(8)))  _Float16 v8h;
typedef __attribute__((ext_vector_type(8)))  float    v8f;

#define EPS_BN 1e-5f

// Async copy of 16 bytes global -> LDS (tracked by ASYNCcnt, no VGPR data).
__device__ __forceinline__ void async_copy_b128(unsigned ldsOff,
                                                const void* gptr) {
  asm volatile("global_load_async_to_lds_b128 %0, %1, off"
               :: "v"(ldsOff), "v"(gptr)
               : "memory");
}
__device__ __forceinline__ void wait_asynccnt0() {
  asm volatile("s_wait_asynccnt 0x0" ::: "memory");
}

// ---------------------------------------------------------------------------
// Farthest point sampling: one block per batch; points + running min-dist in
// LDS (64KB of the 320KB WGP pool); S sequential argmax rounds.
// ---------------------------------------------------------------------------
#define FPS_NMAX 4096
__global__ void fps_kernel(const float* __restrict__ xyz, int* __restrict__ fidx,
                           int Np, int S) {
  __shared__ float sx[FPS_NMAX], sy[FPS_NMAX], sz[FPS_NMAX], sd[FPS_NMAX];
  __shared__ float rv[256];
  __shared__ int   ri[256];
  __shared__ int   sFar;
  const int b = blockIdx.x;
  const int tid = threadIdx.x;
  const float* p = xyz + (size_t)b * Np * 3;
  for (int i = tid; i < Np; i += 256) {
    sx[i] = p[i * 3 + 0];
    sy[i] = p[i * 3 + 1];
    sz[i] = p[i * 3 + 2];
    sd[i] = 1e10f;
  }
  if (tid == 0) sFar = 0;
  __syncthreads();
  for (int it = 0; it < S; ++it) {
    const int far = sFar;
    if (tid == 0) fidx[b * S + it] = far;
    const float cx = sx[far], cy = sy[far], cz = sz[far];
    float best = -1.0f;
    int bi = 0;
    for (int i = tid; i < Np; i += 256) {
      const float dx = sx[i] - cx, dy = sy[i] - cy, dz = sz[i] - cz;
      float d = dx * dx + dy * dy + dz * dz;
      const float old = sd[i];
      d = d < old ? d : old;
      sd[i] = d;
      if (d > best) { best = d; bi = i; }
    }
    rv[tid] = best; ri[tid] = bi;
    __syncthreads();
    for (int s = 128; s > 0; s >>= 1) {
      if (tid < s) {
        if (rv[tid + s] > rv[tid] ||
            (rv[tid + s] == rv[tid] && ri[tid + s] < ri[tid])) {
          rv[tid] = rv[tid + s];
          ri[tid] = ri[tid + s];
        }
      }
      __syncthreads();
    }
    if (tid == 0) sFar = ri[0];
    __syncthreads();
  }
}

__global__ void gather_centroids(const float* __restrict__ xyz,
                                 const int* __restrict__ fidx,
                                 float* __restrict__ nxyz,
                                 int Np, int S, int total) {
  const int t = blockIdx.x * 256 + threadIdx.x;
  if (t >= total) return;
  const int b = t / S;
  const int i = fidx[t];
  const float* p = xyz + ((size_t)b * Np + i) * 3;
  nxyz[t * 3 + 0] = p[0];
  nxyz[t * 3 + 1] = p[1];
  nxyz[t * 3 + 2] = p[2];
}

__global__ void ball_query(const float* __restrict__ xyz,
                           const float* __restrict__ nxyz,
                           int* __restrict__ idx,
                           int Np, int S, int ns, float r2, int total) {
  const int t = blockIdx.x * 256 + threadIdx.x;
  if (t >= total) return;
  const int b = t / S;
  const float* p = xyz + (size_t)b * Np * 3;
  const float cx = nxyz[t * 3 + 0], cy = nxyz[t * 3 + 1], cz = nxyz[t * 3 + 2];
  int* o = idx + (size_t)t * ns;
  int cnt = 0, first = 0;
  bool havef = false;
  for (int i = 0; i < Np && cnt < ns; ++i) {
    const float dx = p[i * 3 + 0] - cx;
    const float dy = p[i * 3 + 1] - cy;
    const float dz = p[i * 3 + 2] - cz;
    if (dx * dx + dy * dy + dz * dz < r2) {
      if (!havef) { first = i; havef = true; }
      o[cnt++] = i;
    }
  }
  for (int j = cnt; j < ns; ++j) o[j] = first;
}

__global__ void group_sa1(const float* __restrict__ xyz,
                          const float* __restrict__ nxyz,
                          const int* __restrict__ idx,
                          half_t* __restrict__ X,
                          int Np, int S, int ns, int total) {
  const int t = blockIdx.x * 256 + threadIdx.x;
  if (t >= total) return;
  const int c = t / ns;
  const int b = c / S;
  const int i = idx[t];
  const float* p = xyz + ((size_t)b * Np + i) * 3;
  const float* q = nxyz + (size_t)c * 3;
  half_t* row = X + (size_t)t * 32;
  row[0] = (half_t)(p[0] - q[0]);
  row[1] = (half_t)(p[1] - q[1]);
  row[2] = (half_t)(p[2] - q[2]);
  for (int k = 3; k < 32; ++k) row[k] = (half_t)0.0f;
}

__global__ void group_sa2(const float* __restrict__ xyz1,
                          const float* __restrict__ nxyz2,
                          const int* __restrict__ idx,
                          const half_t* __restrict__ f1,
                          half_t* __restrict__ X,
                          int Np, int S, int ns, int C, int ldk, int total) {
  const int t = blockIdx.x * 256 + threadIdx.x;
  if (t >= total) return;
  const int c = t / ns;
  const int b = c / S;
  const int i = idx[t];
  const float* p = xyz1 + ((size_t)b * Np + i) * 3;
  const float* q = nxyz2 + (size_t)c * 3;
  const half_t* f = f1 + ((size_t)b * Np + i) * C;
  half_t* row = X + (size_t)t * ldk;
  row[0] = (half_t)(p[0] - q[0]);
  row[1] = (half_t)(p[1] - q[1]);
  row[2] = (half_t)(p[2] - q[2]);
  for (int k = 0; k < C; ++k) row[3 + k] = f[k];
  for (int k = 3 + C; k < ldk; ++k) row[k] = (half_t)0.0f;
}

__global__ void build_sa3(const float* __restrict__ nxyz2,
                          const half_t* __restrict__ f2,
                          half_t* __restrict__ X,
                          int C, int ldk, int total) {
  const int t = blockIdx.x * 256 + threadIdx.x;
  if (t >= total) return;
  const float* q = nxyz2 + (size_t)t * 3;
  const half_t* f = f2 + (size_t)t * C;
  half_t* row = X + (size_t)t * ldk;
  row[0] = (half_t)q[0];
  row[1] = (half_t)q[1];
  row[2] = (half_t)q[2];
  for (int k = 0; k < C; ++k) row[3 + k] = f[k];
  for (int k = 3 + C; k < ldk; ++k) row[k] = (half_t)0.0f;
}

// fp32 weights (N x K) -> fp16, zero-padded to Npad x ldk.
__global__ void convert_w(const float* __restrict__ W, half_t* __restrict__ Wh,
                          int N, int K, int Npad, int ldk) {
  const int t = blockIdx.x * 256 + threadIdx.x;
  const int tot = Npad * ldk;
  if (t >= tot) return;
  const int n = t / ldk;
  const int k = t % ldk;
  Wh[t] = (n < N && k < K) ? (half_t)W[(size_t)n * K + k] : (half_t)0.0f;
}

__global__ void fill_f32(float* __restrict__ p, int n) {
  const int t = blockIdx.x * 256 + threadIdx.x;
  if (t < n) p[t] = 0.0f;
}

// ---------------------------------------------------------------------------
// WMMA GEMM: out[rows x N] = X[rows x ldk] * Wt[N x ldk]^T (+bias)(+relu).
// Block: 256 threads (8 wave32), 128 rows x (NT*16) cols.
// The block's contiguous weight strip (NT*16 rows x ldk halves) is staged to
// LDS with global_load_async_to_lds_b128; chunk count is always a multiple
// of 256 (ldk % 32 == 0), so the staging loop is EXEC-uniform.
// Each wave then holds one A fragment per K-step (ISA 7.12.2 layout: lane
// m<16 holds row m K{0..7,16..23}; lane m+16 holds K{8..15,24..31}) and
// feeds NT accumulators, reusing A across NT column tiles.
// Pad columns (N <= c < ldo) store zeros so the next layer's K-pad is clean.
// ---------------------------------------------------------------------------
template <bool RELU, bool BIAS, typename OutT, int NT>
__global__ __launch_bounds__(256) void wmma_gemm(
    const half_t* __restrict__ X, const half_t* __restrict__ Wt,
    const float* __restrict__ bias, OutT* __restrict__ out,
    int rows, int ldk, int N, int ldo) {
  extern __shared__ half_t sB[];   // NT*16 x ldk halves
  const int tid = threadIdx.x;
  const int nBase = blockIdx.y * (16 * NT);

  // Stage the block's weight strip into LDS (async, ASYNCcnt-tracked).
  {
    const half_t* gB = Wt + (size_t)nBase * ldk;
    const int chunks = 2 * NT * ldk;               // 16B chunks, % 256 == 0
    for (int t = tid; t < chunks; t += 256) {
      const unsigned ldsOff = (unsigned)(size_t)(&sB[t * 8]);
      async_copy_b128(ldsOff, gB + (size_t)t * 8);
    }
    wait_asynccnt0();
  }
  __syncthreads();

  const int lane   = tid & 31;
  const int wave   = tid >> 5;
  const int mTile  = blockIdx.x * 128 + wave * 16;
  if (mTile >= rows) return;       // wave-uniform; no barriers after this
  const int lanelo = lane & 15;
  const int hi     = lane >> 4;

  const half_t* aBase = X + (size_t)(mTile + lanelo) * ldk + hi * 8;
  const half_t* bBase = &sB[(size_t)lanelo * ldk + hi * 8];

  v8f acc[NT];
#pragma unroll
  for (int nt = 0; nt < NT; ++nt) acc[nt] = (v8f){};

  for (int k0 = 0; k0 < ldk; k0 += 32) {
    union { v16h v; v8h h[2]; } a;
    a.h[0] = *(const v8h*)(aBase + k0);
    a.h[1] = *(const v8h*)(aBase + k0 + 16);
    __builtin_prefetch(aBase + k0 + 32, 0, 1);     // global_prefetch_b8
#pragma unroll
    for (int nt = 0; nt < NT; ++nt) {
      union { v16h v; v8h h[2]; } b;
      const half_t* bp = bBase + (size_t)nt * 16 * ldk + k0;   // ds_load_b128
      b.h[0] = *(const v8h*)bp;
      b.h[1] = *(const v8h*)(bp + 16);
      acc[nt] = __builtin_amdgcn_wmma_f32_16x16x32_f16(
          /*neg_a=*/false, a.v, /*neg_b=*/false, b.v,
          /*c_mod=*/(short)0, acc[nt], /*reuse_a=*/false, /*reuse_b=*/false);
    }
  }

  // C/D layout: VGPR i -> row (i + 8*hi), col lanelo.
#pragma unroll
  for (int nt = 0; nt < NT; ++nt) {
    const int c = nBase + nt * 16 + lanelo;
    float bv = 0.0f;
    if (BIAS && c < N) bv = bias[c];
    OutT* orow = out + (size_t)(mTile + hi * 8) * ldo + c;
#pragma unroll
    for (int i = 0; i < 8; ++i) {
      if (c < N) {
        float v = acc[nt][i] + bv;
        if (RELU && v < 0.0f) v = 0.0f;
        orow[(size_t)i * ldo] = (OutT)v;
      } else if (c < ldo) {
        orow[(size_t)i * ldo] = (OutT)0.0f;
      }
    }
  }
}

// Per-channel sum / sum-of-squares with fp32 global atomics.
__global__ void stats_kernel(const half_t* __restrict__ Z, int rows, int N,
                             int ld, float* __restrict__ sum,
                             float* __restrict__ sumsq, int rowsPerBlock) {
  const int c = blockIdx.x * 64 + threadIdx.x;
  if (c >= N) return;
  const int r0 = blockIdx.y * rowsPerBlock;
  int r1 = r0 + rowsPerBlock;
  if (r1 > rows) r1 = rows;
  float s = 0.0f, ss = 0.0f;
  for (int r = r0; r < r1; ++r) {
    const float v = (float)Z[(size_t)r * ld + c];
    s += v;
    ss += v * v;
  }
  atomicAdd(&sum[c], s);
  atomicAdd(&sumsq[c], ss);
}

// In-place batchnorm (stats over all rows) + ReLU.
__global__ void bn_relu_kernel(half_t* __restrict__ Z, int rows, int N, int ld,
                               const float* __restrict__ sum,
                               const float* __restrict__ sumsq,
                               const float* __restrict__ gamma,
                               const float* __restrict__ beta, float invCnt) {
  const size_t t = (size_t)blockIdx.x * 256 + threadIdx.x;
  const size_t tot = (size_t)rows * N;
  if (t >= tot) return;
  const int c = (int)(t % N);
  const size_t r = t / N;
  const float m = sum[c] * invCnt;
  const float var = sumsq[c] * invCnt - m * m;
  const float z = (float)Z[r * ld + c];
  const float y = (z - m) * rsqrtf(var + EPS_BN) * gamma[c] + beta[c];
  Z[r * ld + c] = (half_t)(y > 0.0f ? y : 0.0f);
}

// Max over groups of g consecutive rows.
__global__ void maxpool_kernel(const half_t* __restrict__ in,
                               half_t* __restrict__ out,
                               int orows, int g, int C, int ldin, int ldo) {
  const int t = blockIdx.x * 256 + threadIdx.x;
  const int tot = orows * C;
  if (t >= tot) return;
  const int c = t % C;
  const int o = t / C;
  const half_t* p = in + (size_t)o * g * ldin + c;
  float m = (float)p[0];
  for (int j = 1; j < g; ++j) {
    const float v = (float)p[(size_t)j * ldin];
    if (v > m) m = v;
  }
  out[(size_t)o * ldo + c] = (half_t)m;
}

// ---------------------------------------------------------------------------
// Host-side layer driver (launches only; graph-capture safe).
// ---------------------------------------------------------------------------
static void mlp_bn_layer(hipStream_t stream, const half_t* Xin, half_t* Xout,
                         const float* W, const float* bias, const float* gamma,
                         const float* beta, int rows, int K, int ldk, int N,
                         int ldo, half_t* Wh, float* stats) {
  // ldo is a multiple of 64 for every BN layer -> NT=4 exact coverage.
  const int wtot = ldo * ldk;
  convert_w<<<(wtot + 255) / 256, 256, 0, stream>>>(W, Wh, N, K, ldo, ldk);
  fill_f32<<<(2 * N + 255) / 256, 256, 0, stream>>>(stats, 2 * N);
  dim3 gg((rows + 127) / 128, ldo / 64);
  const size_t shBytes = (size_t)4 * 16 * ldk * sizeof(half_t);  // <= 64KB
  wmma_gemm<false, true, half_t, 4>
      <<<gg, 256, shBytes, stream>>>(Xin, Wh, bias, Xout, rows, ldk, N, ldo);
  dim3 gs((N + 63) / 64, (rows + 1023) / 1024);
  stats_kernel<<<gs, 64, 0, stream>>>(Xout, rows, N, ldo, stats, stats + N, 1024);
  const size_t tot = (size_t)rows * N;
  bn_relu_kernel<<<(unsigned)((tot + 255) / 256), 256, 0, stream>>>(
      Xout, rows, N, ldo, stats, stats + N, gamma, beta, 1.0f / (float)rows);
}

extern "C" void kernel_launch(void* const* d_in, const int* in_sizes, int n_in,
                              void* d_out, int out_size, void* d_ws,
                              size_t ws_size, hipStream_t stream) {
  (void)in_sizes; (void)n_in; (void)out_size; (void)ws_size;

  // Input order: setup_inputs() dict insertion order:
  //   points, sa1[3x{W,b,gamma,beta}], sa2[...], sa3[...], fc[3x{W,b}].
  const float* points = (const float*)d_in[0];
  const int base_sa1 = 1, base_sa2 = 13, base_sa3 = 25, base_fc = 37;
  auto saW = [&](int base, int l) { return (const float*)d_in[base + l * 4 + 0]; };
  auto saB = [&](int base, int l) { return (const float*)d_in[base + l * 4 + 1]; };
  auto saG = [&](int base, int l) { return (const float*)d_in[base + l * 4 + 2]; };
  auto saBt = [&](int base, int l) { return (const float*)d_in[base + l * 4 + 3]; };
  const float* fcW[3] = {(const float*)d_in[base_fc + 0],
                         (const float*)d_in[base_fc + 2],
                         (const float*)d_in[base_fc + 4]};
  const float* fcB[3] = {(const float*)d_in[base_fc + 1],
                         (const float*)d_in[base_fc + 3],
                         (const float*)d_in[base_fc + 5]};

  // Workspace carve-up.
  char* ws = (char*)d_ws;
  size_t off = 0;
  auto alloc = [&](size_t bytes) -> char* {
    char* p = ws + off;
    off = (off + bytes + 255) & ~(size_t)255;
    return p;
  };
  int*    fidx1 = (int*)alloc((size_t)16 * 512 * 4);
  float*  nxyz1 = (float*)alloc((size_t)16 * 512 * 3 * 4);
  int*    idx1  = (int*)alloc((size_t)16 * 512 * 32 * 4);
  int*    fidx2 = (int*)alloc((size_t)16 * 128 * 4);
  float*  nxyz2 = (float*)alloc((size_t)16 * 128 * 3 * 4);
  int*    idx2  = (int*)alloc((size_t)16 * 128 * 64 * 4);
  half_t* Wh    = (half_t*)alloc((size_t)2 * 1024 * 1024);   // 2MB weights
  float*  stats = (float*)alloc((size_t)2 * 1024 * 4);       // sum + sumsq
  half_t* f1    = (half_t*)alloc((size_t)8192 * 128 * 2);
  half_t* f2    = (half_t*)alloc((size_t)2048 * 256 * 2);
  half_t* XA    = (half_t*)alloc((size_t)262144 * 128 * 2);  // 64MB ping
  half_t* XB    = (half_t*)alloc((size_t)262144 * 128 * 2);  // 64MB pong

  // ---- SA1: 4096 pts -> 512 centroids, ns=32, MLP [3->64->64->128] ----
  fps_kernel<<<16, 256, 0, stream>>>(points, fidx1, 4096, 512);
  gather_centroids<<<(16 * 512 + 255) / 256, 256, 0, stream>>>(
      points, fidx1, nxyz1, 4096, 512, 16 * 512);
  ball_query<<<(16 * 512 + 255) / 256, 256, 0, stream>>>(
      points, nxyz1, idx1, 4096, 512, 32, 0.2f * 0.2f, 16 * 512);
  group_sa1<<<(262144 + 255) / 256, 256, 0, stream>>>(
      points, nxyz1, idx1, XA, 4096, 512, 32, 262144);
  mlp_bn_layer(stream, XA, XB, saW(base_sa1, 0), saB(base_sa1, 0),
               saG(base_sa1, 0), saBt(base_sa1, 0), 262144, 3, 32, 64, 64, Wh, stats);
  mlp_bn_layer(stream, XB, XA, saW(base_sa1, 1), saB(base_sa1, 1),
               saG(base_sa1, 1), saBt(base_sa1, 1), 262144, 64, 64, 64, 64, Wh, stats);
  mlp_bn_layer(stream, XA, XB, saW(base_sa1, 2), saB(base_sa1, 2),
               saG(base_sa1, 2), saBt(base_sa1, 2), 262144, 64, 64, 128, 128, Wh, stats);
  maxpool_kernel<<<(8192 * 128 + 255) / 256, 256, 0, stream>>>(
      XB, f1, 8192, 32, 128, 128, 128);

  // ---- SA2: 512 -> 128 centroids, ns=64, MLP [131->128->128->256] ----
  fps_kernel<<<16, 256, 0, stream>>>(nxyz1, fidx2, 512, 128);
  gather_centroids<<<(2048 + 255) / 256, 256, 0, stream>>>(
      nxyz1, fidx2, nxyz2, 512, 128, 2048);
  ball_query<<<(2048 + 255) / 256, 256, 0, stream>>>(
      nxyz1, nxyz2, idx2, 512, 128, 64, 0.4f * 0.4f, 2048);
  group_sa2<<<(131072 + 255) / 256, 256, 0, stream>>>(
      nxyz1, nxyz2, idx2, f1, XA, 512, 128, 64, 128, 160, 131072);
  mlp_bn_layer(stream, XA, XB, saW(base_sa2, 0), saB(base_sa2, 0),
               saG(base_sa2, 0), saBt(base_sa2, 0), 131072, 131, 160, 128, 128, Wh, stats);
  mlp_bn_layer(stream, XB, XA, saW(base_sa2, 1), saB(base_sa2, 1),
               saG(base_sa2, 1), saBt(base_sa2, 1), 131072, 128, 128, 128, 128, Wh, stats);
  mlp_bn_layer(stream, XA, XB, saW(base_sa2, 2), saB(base_sa2, 2),
               saG(base_sa2, 2), saBt(base_sa2, 2), 131072, 128, 128, 256, 256, Wh, stats);
  maxpool_kernel<<<(2048 * 256 + 255) / 256, 256, 0, stream>>>(
      XB, f2, 2048, 64, 256, 256, 256);

  // ---- SA3: global MLP [259->256->512->1024] over 16*128 rows ----
  build_sa3<<<(2048 + 255) / 256, 256, 0, stream>>>(nxyz2, f2, XA, 256, 288, 2048);
  mlp_bn_layer(stream, XA, XB, saW(base_sa3, 0), saB(base_sa3, 0),
               saG(base_sa3, 0), saBt(base_sa3, 0), 2048, 259, 288, 256, 256, Wh, stats);
  mlp_bn_layer(stream, XB, XA, saW(base_sa3, 1), saB(base_sa3, 1),
               saG(base_sa3, 1), saBt(base_sa3, 1), 2048, 256, 256, 512, 512, Wh, stats);
  mlp_bn_layer(stream, XA, XB, saW(base_sa3, 2), saB(base_sa3, 2),
               saG(base_sa3, 2), saBt(base_sa3, 2), 2048, 512, 512, 1024, 1024, Wh, stats);
  maxpool_kernel<<<(16 * 1024 + 255) / 256, 256, 0, stream>>>(
      XB, XA, 16, 128, 1024, 1024, 1024);   // f3: 16 x 1024 in XA

  // ---- FC head: 1024->512 relu, 512->256 relu, 256->40 ----
  // fc1: ldk=1024 -> NT=2 keeps the LDS strip at 64KB.
  convert_w<<<(512 * 1024 + 255) / 256, 256, 0, stream>>>(fcW[0], Wh, 512, 1024, 512, 1024);
  wmma_gemm<true, true, half_t, 2>
      <<<dim3(1, 512 / 32), 256, (size_t)2 * 16 * 1024 * 2, stream>>>(
          XA, Wh, fcB[0], XB, 16, 1024, 512, 512);
  convert_w<<<(256 * 512 + 255) / 256, 256, 0, stream>>>(fcW[1], Wh, 256, 512, 256, 512);
  wmma_gemm<true, true, half_t, 4>
      <<<dim3(1, 256 / 64), 256, (size_t)4 * 16 * 512 * 2, stream>>>(
          XB, Wh, fcB[1], XA, 16, 512, 256, 256);
  // fc3: N=40; stage 64 padded weight rows, store only c<40 (ldo=40).
  convert_w<<<(64 * 256 + 255) / 256, 256, 0, stream>>>(fcW[2], Wh, 40, 256, 64, 256);
  wmma_gemm<false, true, float, 4>
      <<<dim3(1, 1), 256, (size_t)4 * 16 * 256 * 2, stream>>>(
          XA, Wh, fcB[2], (float*)d_out, 16, 256, 40, 40);
}
